// MultiHeadAttention_81509889344259
// MI455X (gfx1250) — matrix operand order repaired and checked by reading, not compiled, not run on hardware
//
#include <hip/hip_runtime.h>

// ---------------------------------------------------------------------------
// MHA forward for MI455X (gfx1250, wave32, WMMA 16x16x32 bf16)
// B=4, S=2048, D=1024, H=16, HD=64
// ---------------------------------------------------------------------------

typedef __bf16 bf16_t;
typedef bf16_t v16bf __attribute__((ext_vector_type(16)));
typedef float  v8f   __attribute__((ext_vector_type(8)));

constexpr int Bn  = 4;
constexpr int Sn  = 2048;
constexpr int Dn  = 1024;
constexpr int Hn  = 16;
constexpr int HDn = 64;
constexpr int N3  = 3 * Dn;   // 3072

// ---- helpers --------------------------------------------------------------

__device__ __forceinline__ unsigned short f2bf(float f) {
    union { float f; unsigned u; } c; c.f = f;
    unsigned r = c.u + 0x7FFFu + ((c.u >> 16) & 1u);   // round-to-nearest-even
    return (unsigned short)(r >> 16);
}

__device__ __forceinline__ v16bf as_v16bf(uint4 a, uint4 b) {
    union { uint4 u[2]; v16bf v; } x;
    x.u[0] = a; x.u[1] = b;
    return x.v;
}

__device__ __forceinline__ v8f wmma_bf16(v16bf a, v16bf b, v8f c) {
    // (neg_a, A, neg_b, B, c_mod, C, reuse_a, reuse_b)
    return __builtin_amdgcn_wmma_f32_16x16x32_bf16(false, a, false, b,
                                                   (short)0, c, false, false);
}

// A fragment: 16(M) x 32(K) bf16 tile, row-major source with leading dim ld.
// Lane layout (ISA 7.12.2): M = lane%16, k0 = (lane<16)?0:8,
//   dwords 0-3 -> K=k0..k0+7, dwords 4-7 -> K=k0+16..k0+23  (two b128 loads)
__device__ __forceinline__ v16bf load_a_frag(const unsigned short* base,
                                             size_t ld, int row0, int k0col,
                                             int lane) {
    int m  = lane & 15;
    int k0 = (lane >> 4) ? 8 : 0;
    const unsigned short* p = base + (size_t)(row0 + m) * ld + k0col + k0;
    return as_v16bf(*(const uint4*)p, *(const uint4*)(p + 16));
}

// B fragment: 32(K) x 16(N) bf16 tile, row-major source with leading dim ld.
// Lane layout: N = lane%16, k0 = (lane<16)?0:8, elems 0-7 -> K=k0..k0+7,
//   elems 8-15 -> K=k0+16..k0+23 (strided gather)
__device__ __forceinline__ v16bf load_b_frag(const unsigned short* base,
                                             size_t ld, int krow0, int n0,
                                             int lane) {
    const int k0 = (lane >> 4) ? 8 : 0;
    const unsigned short* p = base + (size_t)(krow0 + k0) * ld + n0 + (lane & 15);
    union { unsigned short s[16]; v16bf v; } x;
    #pragma unroll
    for (int i = 0; i < 8; ++i) {
        x.s[i]     = p[(size_t)i * ld];
        x.s[8 + i] = p[(size_t)(16 + i) * ld];
    }
    return x.v;
}

// fragment set for a 32M x 64N wave tile over one 32-wide K chunk
struct Frags {
    v16bf a[2];
    v16bf b[4];
};

__device__ __forceinline__ void load_frags(Frags& F,
                                           const unsigned short* Am, size_t lda,
                                           const unsigned short* Bm, size_t ldb,
                                           int row0, int col0, int kc, int lane) {
    F.a[0] = load_a_frag(Am, lda, row0,      kc, lane);
    F.a[1] = load_a_frag(Am, lda, row0 + 16, kc, lane);
    #pragma unroll
    for (int f = 0; f < 4; ++f)
        F.b[f] = load_b_frag(Bm, ldb, kc, col0 + 16 * f, lane);
}

__device__ __forceinline__ void mma_frags(v8f (&acc)[2][4], const Frags& F) {
    #pragma unroll
    for (int i = 0; i < 2; ++i)
        #pragma unroll
        for (int f = 0; f < 4; ++f)
            acc[i][f] = wmma_bf16(F.a[i], F.b[f], acc[i][f]);
}

// ---- stage 1: f32 -> bf16 conversion --------------------------------------

__global__ void convert_f32_bf16(const float* __restrict__ in,
                                 unsigned short* __restrict__ out,
                                 size_t n4) {
    size_t i = (size_t)blockIdx.x * blockDim.x + threadIdx.x;
    size_t stride = (size_t)gridDim.x * blockDim.x;
    for (; i < n4; i += stride) {
        float4 v = ((const float4*)in)[i];
        uint2 o;
        o.x = (unsigned)f2bf(v.x) | ((unsigned)f2bf(v.y) << 16);
        o.y = (unsigned)f2bf(v.z) | ((unsigned)f2bf(v.w) << 16);
        ((uint2*)out)[i] = o;
    }
}

// ---- stage 2: QKV GEMM: [8192,1024] x [1024,3072] + bias -------------------
// Block = 256 thr = 8 waves (2 along M x 4 along N). Wave tile = 32M x 64N,
// K unrolled x2 with ping-pong fragment sets (no register copies).
// Epilogue scatters Q,V as [B,H,S,64] bf16 and K transposed as [B,H,64,S].

__global__ __launch_bounds__(256)
void qkv_gemm(const unsigned short* __restrict__ xb,
              const unsigned short* __restrict__ wA,
              const float* __restrict__ bias,
              unsigned short* __restrict__ qb,
              unsigned short* __restrict__ kTb,
              unsigned short* __restrict__ vb) {
    const int tid  = threadIdx.x;
    const int lane = tid & 31;
    const int wave = tid >> 5;
    const int wm   = wave & 1;
    const int wn   = wave >> 1;

    const int row0 = blockIdx.y * 64 + wm * 32;          // M (two 16-row tiles)
    const int col0 = blockIdx.x * 256 + wn * 64;         // N (multiple of 64)

    v8f acc[2][4] = {};
    Frags F0, F1;
    load_frags(F0, xb, Dn, wA, N3, row0, col0, 0, lane);

    for (int kc = 0; kc < Dn; kc += 64) {
        load_frags(F1, xb, Dn, wA, N3, row0, col0, kc + 32, lane);
        mma_frags(acc, F0);
        if (kc + 64 < Dn)
            load_frags(F0, xb, Dn, wA, N3, row0, col0, kc + 64, lane);
        mma_frags(acc, F1);
    }

    const int half = lane >> 4;
    const int col  = lane & 15;
    #pragma unroll
    for (int i = 0; i < 2; ++i) {
        const int rowb = row0 + i * 16;
        #pragma unroll
        for (int f = 0; f < 4; ++f) {
            const int n0  = col0 + 16 * f;
            const int n   = n0 + col;
            const float bv = bias[n];
            const int g64 = n0 >> 6;       // which 64-wide column group
            const int mat = g64 >> 4;      // 0=q, 1=k, 2=v
            const int h   = g64 & 15;
            const int d   = n & 63;
            #pragma unroll
            for (int r = 0; r < 8; ++r) {
                const int row = rowb + r + 8 * half;     // token index [0,8192)
                const int bI  = row >> 11;               // /2048
                const int s   = row & 2047;
                const unsigned short val = f2bf(acc[i][f][r] + bv);
                if (mat == 0)
                    qb[(((size_t)bI * Hn + h) * Sn + s) * HDn + d] = val;
                else if (mat == 1)
                    kTb[(((size_t)bI * Hn + h) * HDn + d) * Sn + s] = val;
                else
                    vb[(((size_t)bI * Hn + h) * Sn + s) * HDn + d] = val;
            }
        }
    }
}

// ---- stage 3: causal flash attention --------------------------------------
// 4 waves/block; each wave owns one 16-query tile and streams key blocks of
// 32 with online softmax. P is re-shaped C-layout -> A-layout via LDS.

__global__ __launch_bounds__(128)
void flash_attn(const unsigned short* __restrict__ qb,
                const unsigned short* __restrict__ kTb,
                const unsigned short* __restrict__ vb,
                unsigned short* __restrict__ aob) {
    __shared__ __align__(16) unsigned short Pl[4][16][32];

    const int tid  = threadIdx.x;
    const int lane = tid & 31;
    const int wave = tid >> 5;
    const int half = lane >> 4;
    const int col  = lane & 15;

    const int bI = blockIdx.z;
    const int h  = blockIdx.y;
    const int q0 = (blockIdx.x * 4 + wave) * 16;

    const int bh = bI * Hn + h;
    const size_t qvbase = (size_t)bh * Sn * HDn;         // q and v layout
    const size_t ktbase = (size_t)bh * HDn * Sn;         // k transposed

    // Q fragments for both 32-wide K-chunks of HD=64
    v16bf aq[2];
    aq[0] = load_a_frag(qb + qvbase, HDn, q0, 0,  lane);
    aq[1] = load_a_frag(qb + qvbase, HDn, q0, 32, lane);

    v8f O[4] = {};
    float m_r[8], l_r[8];
    #pragma unroll
    for (int r = 0; r < 8; ++r) { m_r[r] = -__builtin_inff(); l_r[r] = 0.0f; }

    for (int kb = 0; kb < q0 + 16; kb += 32) {
        // scores: S_tile[16q x 32k] = Q(16x64) * K^T(64x32), split K dim
        v16bf bk00 = load_b_frag(kTb + ktbase, Sn, 0,  kb,      lane);
        v16bf bk01 = load_b_frag(kTb + ktbase, Sn, 32, kb,      lane);
        v16bf bk10 = load_b_frag(kTb + ktbase, Sn, 0,  kb + 16, lane);
        v16bf bk11 = load_b_frag(kTb + ktbase, Sn, 32, kb + 16, lane);
        v8f sc[2];
        {
            v8f z0 = {}; z0 = wmma_bf16(aq[0], bk00, z0);
            sc[0] = wmma_bf16(aq[1], bk01, z0);
            v8f z1 = {}; z1 = wmma_bf16(aq[0], bk10, z1);
            sc[1] = wmma_bf16(aq[1], bk11, z1);
        }

        // online softmax in C-fragment layout
        #pragma unroll
        for (int r = 0; r < 8; ++r) {
            const int qrow = q0 + r + 8 * half;
            float mx = -__builtin_inff();
            #pragma unroll
            for (int t = 0; t < 2; ++t) {
                float sv = sc[t][r] * 0.125f;            // 1/sqrt(64)
                const int key = kb + 16 * t + col;
                if (key > qrow) sv = -__builtin_inff();  // causal mask
                sc[t][r] = sv;
                mx = fmaxf(mx, sv);
            }
            #pragma unroll
            for (int off = 1; off < 16; off <<= 1)
                mx = fmaxf(mx, __shfl_xor(mx, off, 16)); // row max (16 lanes)
            const float mnew  = fmaxf(m_r[r], mx);
            const float alpha = __expf(m_r[r] - mnew);
            float rs = 0.0f;
            #pragma unroll
            for (int t = 0; t < 2; ++t) {
                const float p = __expf(sc[t][r] - mnew);
                Pl[wave][r + 8 * half][16 * t + col] = f2bf(p);
                rs += p;
            }
            #pragma unroll
            for (int off = 1; off < 16; off <<= 1)
                rs += __shfl_xor(rs, off, 16);           // row sum
            l_r[r] = l_r[r] * alpha + rs;
            m_r[r] = mnew;
            #pragma unroll
            for (int f = 0; f < 4; ++f) O[f][r] *= alpha;
        }

        // make the per-wave LDS P tile readable (same-wave DS ordering)
        asm volatile("s_wait_dscnt 0" ::: "memory");

        // P as A-fragment (16q x 32k)
        const int am = lane & 15;
        const int ak = half ? 8 : 0;
        v16bf ap = as_v16bf(*(const uint4*)&Pl[wave][am][ak],
                            *(const uint4*)&Pl[wave][am][ak + 16]);

        // V block: 32 key rows x 64 hd -> four B fragments
        #pragma unroll
        for (int f = 0; f < 4; ++f) {
            v16bf bv = load_b_frag(vb + qvbase, HDn, kb, 16 * f, lane);
            O[f] = wmma_bf16(ap, bv, O[f]);
        }
    }

    // normalize and emit attn output as bf16 [B*S, D]
    #pragma unroll
    for (int r = 0; r < 8; ++r) {
        const float inv = 1.0f / l_r[r];
        const int row = q0 + r + 8 * half;
        const size_t obase = ((size_t)bI * Sn + row) * Dn + (size_t)h * HDn;
        #pragma unroll
        for (int f = 0; f < 4; ++f)
            aob[obase + f * 16 + col] = f2bf(O[f][r] * inv);
    }
}

// ---- stage 4: output projection: [8192,1024] x [1024,1024] + bias ---------

__global__ __launch_bounds__(256)
void proj_gemm(const unsigned short* __restrict__ aob,
               const unsigned short* __restrict__ wP,
               const float* __restrict__ bias,
               float* __restrict__ out) {
    const int tid  = threadIdx.x;
    const int lane = tid & 31;
    const int wave = tid >> 5;
    const int wm   = wave & 1;
    const int wn   = wave >> 1;

    const int row0 = blockIdx.y * 64 + wm * 32;
    const int col0 = blockIdx.x * 256 + wn * 64;

    v8f acc[2][4] = {};
    Frags F0, F1;
    load_frags(F0, aob, Dn, wP, Dn, row0, col0, 0, lane);

    for (int kc = 0; kc < Dn; kc += 64) {
        load_frags(F1, aob, Dn, wP, Dn, row0, col0, kc + 32, lane);
        mma_frags(acc, F0);
        if (kc + 64 < Dn)
            load_frags(F0, aob, Dn, wP, Dn, row0, col0, kc + 64, lane);
        mma_frags(acc, F1);
    }

    const int half = lane >> 4;
    const int col  = lane & 15;
    #pragma unroll
    for (int i = 0; i < 2; ++i) {
        const int rowb = row0 + i * 16;
        #pragma unroll
        for (int f = 0; f < 4; ++f) {
            const int n = col0 + 16 * f + col;
            const float bv = bias[n];
            #pragma unroll
            for (int r = 0; r < 8; ++r) {
                const int row = rowb + r + 8 * half;
                out[(size_t)row * Dn + n] = acc[i][f][r] + bv;
            }
        }
    }
}

// ---------------------------------------------------------------------------

extern "C" void kernel_launch(void* const* d_in, const int* in_sizes, int n_in,
                              void* d_out, int out_size, void* d_ws, size_t ws_size,
                              hipStream_t stream) {
    const float* x      = (const float*)d_in[0];   // [4,2048,1024]
    const float* w_attn = (const float*)d_in[1];   // [1024,3072]
    const float* b_attn = (const float*)d_in[2];   // [3072]
    const float* w_proj = (const float*)d_in[3];   // [1024,1024]
    const float* b_proj = (const float*)d_in[4];   // [1024]
    float* out = (float*)d_out;                    // [4,2048,1024]

    const size_t nX  = (size_t)Bn * Sn * Dn;       // 8,388,608
    const size_t nWA = (size_t)Dn * N3;            // 3,145,728
    const size_t nWP = (size_t)Dn * Dn;            // 1,048,576
    const size_t nHD = (size_t)Bn * Hn * Sn * HDn; // 8,388,608

    // workspace layout (bf16 = 2 bytes), 256B aligned
    size_t off = 0;
    auto take = [&](size_t elems) {
        size_t o = off;
        off = (off + elems * 2 + 255) & ~(size_t)255;
        return o;
    };
    char* ws = (char*)d_ws;
    unsigned short* xb  = (unsigned short*)(ws + take(nX));
    unsigned short* wAb = (unsigned short*)(ws + take(nWA));
    unsigned short* wPb = (unsigned short*)(ws + take(nWP));
    unsigned short* qb  = (unsigned short*)(ws + take(nHD));
    unsigned short* kTb = (unsigned short*)(ws + take(nHD));
    unsigned short* vb  = (unsigned short*)(ws + take(nHD));
    unsigned short* aob = (unsigned short*)(ws + take(nX));
    (void)ws_size; (void)in_sizes; (void)n_in; (void)out_size;

    // stage 1: conversions
    convert_f32_bf16<<<(unsigned)(nX  / 4 / 256), 256, 0, stream>>>(x, xb, nX / 4);
    convert_f32_bf16<<<(unsigned)(nWA / 4 / 256), 256, 0, stream>>>(w_attn, wAb, nWA / 4);
    convert_f32_bf16<<<(unsigned)(nWP / 4 / 256), 256, 0, stream>>>(w_proj, wPb, nWP / 4);

    // stage 2: QKV projection (M=8192, N=3072, K=1024)
    qkv_gemm<<<dim3(N3 / 256, (Bn * Sn) / 64), 256, 0, stream>>>(
        xb, wAb, b_attn, qb, kTb, vb);

    // stage 3: causal flash attention
    flash_attn<<<dim3(Sn / 64, Hn, Bn), 128, 0, stream>>>(qb, kTb, vb, aob);

    // stage 4: output projection (M=8192, N=1024, K=1024)
    proj_gemm<<<dim3(Dn / 256, (Bn * Sn) / 64), 256, 0, stream>>>(
        aob, wPb, b_proj, out);
}
